// multiheaded_74371653697779
// MI455X (gfx1250) — compile-verified, matrix operand order checked by reading, not run on hardware
//
#include <hip/hip_runtime.h>

// ---------------------------------------------------------------------------
// MI455X (gfx1250): Linformer attention + edge linear + LayerNorms.
// All GEMMs -> v_wmma_f32_16x16x32_bf16 (fp32 accumulate), fp32 operands
// converted to bf16 (packed cvt via vector fptrunc) while staging into
// double-buffered LDS. Softmax / LayerNorm in fp32 VALU. wave32.
// ---------------------------------------------------------------------------

typedef __attribute__((ext_vector_type(16))) __bf16 v16bf;
typedef __attribute__((ext_vector_type(2)))  __bf16 v2bf;
typedef __attribute__((ext_vector_type(2)))  float  v2f;
typedef __attribute__((ext_vector_type(8)))  float  v8f;

#define LDP 72   // LDS row stride in bf16 elems (64 data + 8 pad; 144B rows -> clean b128 reads)

// Packed f32x2 -> bf16x2. Vector fptrunc gives the backend a clean shot at
// v_cvt_pk_bf16_f32 instead of a scalar bit-twiddle chain.
__device__ __forceinline__ unsigned pack2(float a, float b) {
  v2f in; in.x = a; in.y = b;
  v2bf out = __builtin_convertvector(in, v2bf);
  unsigned u;
  __builtin_memcpy(&u, &out, 4);
  return u;
}

// 16x32 bf16 WMMA fragment from LDS, CDNA5 16-bit A/B layout:
// lane l<16: row (l&15), K=[0..7],[16..23] (+kk); lanes>=16: K=[8..15],[24..31].
__device__ __forceinline__ v16bf ld_frag(const unsigned short* S, int row, int kk, int lane) {
  const int r  = row + (lane & 15);
  const int ko = kk + ((lane >> 4) << 3);
  const unsigned short* p = S + r * LDP + ko;
  union { uint4 u[2]; v16bf v; } q;
  q.u[0] = *(const uint4*)(p);
  q.u[1] = *(const uint4*)(p + 16);
  return q.v;
}

// ---- staging helpers: fp32 global -> bf16 LDS (per-thread base precomputed) ----
__device__ __forceinline__ void stage_a(unsigned short* dst, const float* src, long rowStep) {
  #pragma unroll
  for (int j = 0; j < 8; ++j) {                       // 128 rows / 16
    const float4 f = *(const float4*)(src + j * rowStep);
    unsigned* d = (unsigned*)(dst + j * 16 * LDP);
    d[0] = pack2(f.x, f.y);
    d[1] = pack2(f.z, f.w);
  }
}

__device__ __forceinline__ void stage_b_n(unsigned short* dst, const float* src, long rowStep) {
  #pragma unroll
  for (int j = 0; j < 4; ++j) {                       // 64 rows / 16
    const float4 f = *(const float4*)(src + j * rowStep);
    unsigned* d = (unsigned*)(dst + j * 16 * LDP);
    d[0] = pack2(f.x, f.y);
    d[1] = pack2(f.z, f.w);
  }
}

// B stored K-major: load contiguous n, scatter-transpose into LDS.
__device__ __forceinline__ void stage_b_t(unsigned short* Sb, const float* src, long rowStep,
                                          int sr, int sc4) {
  #pragma unroll
  for (int j = 0; j < 4; ++j) {                       // 64 k-rows / 16
    const float4 f = *(const float4*)(src + j * rowStep);
    const int k = sr + j * 16;
    const unsigned u01 = pack2(f.x, f.y);
    const unsigned u23 = pack2(f.z, f.w);
    Sb[(sc4 + 0) * LDP + k] = (unsigned short)u01;
    Sb[(sc4 + 1) * LDP + k] = (unsigned short)(u01 >> 16);
    Sb[(sc4 + 2) * LDP + k] = (unsigned short)u23;
    Sb[(sc4 + 3) * LDP + k] = (unsigned short)(u23 >> 16);
  }
}

// Load all fragments for one 64-K chunk, then 8 back-to-back WMMAs.
__device__ __forceinline__ void compute_chunk(const unsigned short* Sa, const unsigned short* Sb,
                                              int wave, int lane, v8f acc[4]) {
  const v16bf a0 = ld_frag(Sa, wave * 16, 0,  lane);
  const v16bf a1 = ld_frag(Sa, wave * 16, 32, lane);
  v16bf b0[4], b1[4];
  #pragma unroll
  for (int t = 0; t < 4; ++t) {
    b0[t] = ld_frag(Sb, t * 16, 0,  lane);
    b1[t] = ld_frag(Sb, t * 16, 32, lane);
  }
  #pragma unroll
  for (int t = 0; t < 4; ++t)
    acc[t] = __builtin_amdgcn_wmma_f32_16x16x32_bf16(false, a0, false, b0[t], (short)0, acc[t], false, false);
  #pragma unroll
  for (int t = 0; t < 4; ++t)
    acc[t] = __builtin_amdgcn_wmma_f32_16x16x32_bf16(false, a1, false, b1[t], (short)0, acc[t], false, false);
}

// Generic GEMM: C(M x Nout) = A(M x Kdim) * B^T (+ bias), A/B fp32 in memory.
// BT=false: B row-major (Nout x Kdim), (n,k) at B[n*ldb+k]   (weights, pK)
// BT=true : B K-major   (Kdim x Nout), (n,k) at B[k*ldb+n]   (Kh/Vh slices, pV^T)
// 256 threads = 8 wave32; tile 128x64; K chunked by 64, double-buffered LDS.
// Requires: M%128==0, Nout%64==0, Kdim%64==0, lda/ldb%4==0.
template<bool BT>
__global__ __launch_bounds__(256) void gemm_bf16_wmma(
    const float* __restrict__ A, int lda,
    const float* __restrict__ B, int ldb,
    const float* __restrict__ bias,
    float* __restrict__ C, int ldc,
    int M, int Nout, int Kdim)
{
  (void)M; (void)Nout;
  __shared__ unsigned short As[2][128 * LDP];
  __shared__ unsigned short Bs[2][64 * LDP];

  const int tid  = threadIdx.x;
  const int lane = tid & 31;
  const int wave = tid >> 5;
  const int n0   = blockIdx.x * 64;    // x = column tiles (fast) -> A panel L2 reuse
  const int row0 = blockIdx.y * 128;

  // staging mapping: thread -> (row sr + j*16, col group sc4)
  const int sr  = tid >> 4;            // 0..15
  const int sc4 = (tid & 15) << 2;     // 0,4,...,60
  const long aStep = (long)16 * lda;
  const long bStep = (long)16 * ldb;
  const float* aP = A + (long)(row0 + sr) * lda + sc4;
  const float* bP = BT ? (B + (long)sr * ldb + n0 + sc4)
                       : (B + (long)(n0 + sr) * ldb + sc4);
  const int aOff = sr * LDP + sc4;
  const long bAdv = BT ? (long)64 * ldb : 64;

  const v8f zero = {0.f, 0.f, 0.f, 0.f, 0.f, 0.f, 0.f, 0.f};
  v8f acc[4] = {zero, zero, zero, zero};

  // prologue: stage chunk 0
  stage_a(&As[0][aOff], aP, aStep);
  if (!BT) stage_b_n(&Bs[0][aOff], bP, bStep);
  else     stage_b_t(Bs[0], bP, bStep, sr, sc4);
  aP += 64; bP += bAdv;
  __syncthreads();

  const int nch = Kdim >> 6;
  for (int c = 0; c < nch; ++c) {
    const int cur = c & 1;
    if (c + 1 < nch) {                 // stage next chunk into the other buffer
      const int nxt = cur ^ 1;
      stage_a(&As[nxt][aOff], aP, aStep);
      if (!BT) stage_b_n(&Bs[nxt][aOff], bP, bStep);
      else     stage_b_t(Bs[nxt], bP, bStep, sr, sc4);
      aP += 64; bP += bAdv;
      __builtin_prefetch(aP, 0, 1);    // global_prefetch_b8 for chunk c+2
    }
    compute_chunk(As[cur], Bs[cur], wave, lane, acc);
    __syncthreads();                   // next staging done + our reads of cur done
  }

  // epilogue: C/D layout -> lane holds col (lane&15), rows mb..mb+7
  const int nl = lane & 15;
  const int mb = wave * 16 + ((lane >> 4) << 3);
  #pragma unroll
  for (int t = 0; t < 4; ++t) {
    const int n = n0 + t * 16 + nl;
    const float bv = bias ? bias[n] : 0.0f;
    #pragma unroll
    for (int v = 0; v < 8; ++v) {
      C[(long)(row0 + mb + v) * ldc + n] = acc[t][v] + bv;
    }
  }
}

// Row softmax over 256 columns, in place, with pre-scale. One block per row.
__global__ __launch_bounds__(256) void softmax256(float* __restrict__ S, float scale) {
  __shared__ float red[256];
  const int tid = threadIdx.x;
  const long base = (long)blockIdx.x * 256;
  const float v = S[base + tid] * scale;
  red[tid] = v;
  __syncthreads();
  for (int s = 128; s > 0; s >>= 1) {
    if (tid < s) red[tid] = fmaxf(red[tid], red[tid + s]);
    __syncthreads();
  }
  const float mx = red[0];
  __syncthreads();
  const float e = __expf(v - mx);
  red[tid] = e;
  __syncthreads();
  for (int s = 128; s > 0; s >>= 1) {
    if (tid < s) red[tid] += red[tid + s];
    __syncthreads();
  }
  S[base + tid] = e / red[0];
}

// In-place LayerNorm over 512 cols + residual add. One block (256 thr) per row.
__global__ __launch_bounds__(256) void ln_residual512(
    float* __restrict__ Y, const float* __restrict__ R,
    const float* __restrict__ g, const float* __restrict__ b)
{
  __shared__ float red[256];
  const int tid = threadIdx.x;
  const long base = (long)blockIdx.x * 512;
  const float v0 = Y[base + tid];
  const float v1 = Y[base + 256 + tid];
  red[tid] = v0 + v1;
  __syncthreads();
  for (int s = 128; s > 0; s >>= 1) { if (tid < s) red[tid] += red[tid + s]; __syncthreads(); }
  const float mu = red[0] * (1.0f / 512.0f);
  __syncthreads();
  const float d0 = v0 - mu, d1 = v1 - mu;
  red[tid] = d0 * d0 + d1 * d1;
  __syncthreads();
  for (int s = 128; s > 0; s >>= 1) { if (tid < s) red[tid] += red[tid + s]; __syncthreads(); }
  const float inv = rsqrtf(red[0] * (1.0f / 512.0f) + 1e-5f);
  Y[base + tid]       = d0 * inv * g[tid]       + b[tid]       + R[base + tid];
  Y[base + 256 + tid] = d1 * inv * g[tid + 256] + b[tid + 256] + R[base + 256 + tid];
}

// ---------------------------------------------------------------------------
extern "C" void kernel_launch(void* const* d_in, const int* in_sizes, int n_in,
                              void* d_out, int out_size, void* d_ws, size_t ws_size,
                              hipStream_t stream) {
  (void)in_sizes; (void)n_in; (void)out_size; (void)ws_size;

  constexpr int N = 8192, E = 131072, C = 512, H = 8, D = 64, KR = 256;

  const float* node = (const float*)d_in[0];
  const float* edge = (const float*)d_in[1];
  const float* Wq   = (const float*)d_in[2];
  const float* bq   = (const float*)d_in[3];
  const float* Wk   = (const float*)d_in[4];
  const float* bk   = (const float*)d_in[5];
  const float* Wv   = (const float*)d_in[6];
  const float* bv   = (const float*)d_in[7];
  const float* We   = (const float*)d_in[8];
  const float* be   = (const float*)d_in[9];
  const float* Ep   = (const float*)d_in[10];
  const float* Fp   = (const float*)d_in[11];
  const float* lng  = (const float*)d_in[12];
  const float* lnb  = (const float*)d_in[13];
  const float* leg  = (const float*)d_in[14];
  const float* leb  = (const float*)d_in[15];

  float* out_x = (float*)d_out;                       // (N, C) pre-LN then final
  float* out_y = out_x + (long)N * C;                 // (E, C) pre-LN then final

  // workspace (fp32): qkv (N x 3C), pK (H,KR,D), pV (H,KR,D), scores (H,N,KR)
  float* qkv    = (float*)d_ws;
  float* pK     = qkv + (long)N * 3 * C;
  float* pV     = pK  + (long)H * KR * D;
  float* scores = pV  + (long)H * KR * D;

  const dim3 blk(256);

  // ---- QKV projections: qkv[:,0:512]=Q, [512:1024]=K, [1024:1536]=V ----
  gemm_bf16_wmma<false><<<dim3(C / 64, N / 128), blk, 0, stream>>>(
      node, C, Wq, C, bq, qkv + 0,     3 * C, N, C, C);
  gemm_bf16_wmma<false><<<dim3(C / 64, N / 128), blk, 0, stream>>>(
      node, C, Wk, C, bk, qkv + C,     3 * C, N, C, C);
  gemm_bf16_wmma<false><<<dim3(C / 64, N / 128), blk, 0, stream>>>(
      node, C, Wv, C, bv, qkv + 2 * C, 3 * C, N, C, C);

  // ---- edge linear: pre-LN result straight into d_out y region ----
  gemm_bf16_wmma<false><<<dim3(C / 64, E / 128), blk, 0, stream>>>(
      edge, C, We, C, be, out_y, C, E, C, C);

  // ---- Linformer projections per head: pK = Ep @ Kh, pV = Fp @ Vh ----
  for (int h = 0; h < H; ++h) {
    gemm_bf16_wmma<true><<<dim3(1, KR / 128), blk, 0, stream>>>(
        Ep + (long)h * KR * N, N,           // A: (KR x N)
        qkv + C + h * D, 3 * C,             // B(n=d,k=node) = Kh[node][d]
        nullptr, pK + (long)h * KR * D, D, KR, D, N);
    gemm_bf16_wmma<true><<<dim3(1, KR / 128), blk, 0, stream>>>(
        Fp + (long)h * KR * N, N,
        qkv + 2 * C + h * D, 3 * C,
        nullptr, pV + (long)h * KR * D, D, KR, D, N);
  }

  // ---- scores = Q @ pK^T : (N x KR) per head ----
  for (int h = 0; h < H; ++h) {
    gemm_bf16_wmma<false><<<dim3(KR / 64, N / 128), blk, 0, stream>>>(
        qkv + h * D, 3 * C,                 // A: Qh (N x D), lda=1536
        pK + (long)h * KR * D, D,           // B: pK (KR x D) row-major
        nullptr, scores + (long)h * N * KR, KR, N, KR, D);
  }

  // ---- softmax over KR with 1/sqrt(D) pre-scale, in place ----
  softmax256<<<H * N, blk, 0, stream>>>(scores, 0.125f);

  // ---- out = attn @ pV : (N x D) per head, written into x columns ----
  for (int h = 0; h < H; ++h) {
    gemm_bf16_wmma<true><<<dim3(1, N / 128), blk, 0, stream>>>(
        scores + (long)h * N * KR, KR,      // A: attn (N x KR)
        pV + (long)h * KR * D, D,           // B(n=d,k=r) = pV[r][d]
        nullptr, out_x + h * D, C, N, D, KR);
  }

  // ---- LayerNorm + residual, in place on d_out ----
  ln_residual512<<<N, blk, 0, stream>>>(out_x, node, lng, lnb);
  ln_residual512<<<E, blk, 0, stream>>>(out_y, edge, leg, leb);
}
